// TitansMemorySystem_62972810494567
// MI455X (gfx1250) — compile-verified, matrix operand order checked by reading, not run on hardware
//
#include <hip/hip_runtime.h>
#include <cstdint>

// ---------------------------------------------------------------------------
// Titans memory system for MI455X (gfx1250, wave32, WMMA bf16 16x16x32).
// All heavy GEMMs (incl. both attention stages) run on v_wmma_f32_16x16x32_bf16.
// GEMM tiles are staged to LDS via GLOBAL_LOAD_ASYNC_TO_LDS_B128 (ASYNCcnt,
// double-buffered) when the toolchain exposes the builtin; otherwise a
// register-staged pipelined copy is used.
// The jax.grad "surprise" branch is multiplied by 0.0 in the reference and is
// skipped entirely.
// ---------------------------------------------------------------------------

typedef __bf16 bf16;
typedef __bf16 bf16x16 __attribute__((ext_vector_type(16)));
typedef float  f32x8   __attribute__((ext_vector_type(8)));
typedef int    v4i     __attribute__((ext_vector_type(4)));

#define HDIM   1024
#define BATCH  4
#define SEQ    4096
#define NTOK   (BATCH * SEQ)   // 16384
#define WKV    512
#define MMEM   4096
#define PVEC   64
#define NHEAD  16
#define DHEAD  64
#define LNEPS  1e-12f

#if defined(__HIP_DEVICE_COMPILE__) && \
    __has_builtin(__builtin_amdgcn_global_load_async_to_lds_b128) && \
    __has_builtin(__builtin_amdgcn_s_wait_asynccnt)
#define HAVE_ASYNC 1
#else
#define HAVE_ASYNC 0
#endif

#if HAVE_ASYNC
// 16B global -> LDS async copy (tracked by ASYNCcnt; no VGPR round trip).
// Builtin signature (from clang diagnostic): (v4i addrspace(1)*, v4i
// addrspace(3)*, imm offset, imm cpol).
typedef __attribute__((address_space(1))) v4i gv4i_t;
typedef __attribute__((address_space(3))) v4i lv4i_t;
__device__ __forceinline__ void async_cp16(const bf16* g, bf16* l) {
  __builtin_amdgcn_global_load_async_to_lds_b128(
      (gv4i_t*)(void*)(g),
      (lv4i_t*)(unsigned long)(unsigned)(uintptr_t)(void*)(l),
      0, 0);
}
#endif

// k-index inside a 16x32 bf16 WMMA fragment for element i (0..15) of `lane`.
// CDNA5 ISA 7.12.2 "16-bit A-Matrix 16x32": two contiguous 8-element runs per
// lane (k = base..base+7 and 16+base..16+base+7; base = 8 for lanes 16-31), so
// the compiler lowers each fragment gather to two ds_load_b128.
__device__ __forceinline__ int frag_k(int i, int lane) {
  int v = i >> 1, p = i & 1;
  return ((v & 4) << 2) + ((lane >> 4) << 3) + ((v & 3) << 1) + p;
}

__device__ __forceinline__ f32x8 wmma_bf16(bf16x16 a, bf16x16 b, f32x8 c) {
  return __builtin_amdgcn_wmma_f32_16x16x32_bf16(false, a, false, b,
                                                 (short)0, c, false, false);
}

// ------------------------------- utilities ---------------------------------

__global__ __launch_bounds__(256) void cvt_f32_bf16(const float* __restrict__ s,
                                                    bf16* __restrict__ d, long n) {
  long i = (long)blockIdx.x * 256 + threadIdx.x;
  if (i < n) d[i] = (bf16)s[i];
}

// dst[c*rows + r] = src[r*cols + c]  (f32 -> bf16 transpose)
__global__ __launch_bounds__(256) void transpose_f32_bf16(const float* __restrict__ s,
                                                          bf16* __restrict__ d,
                                                          int rows, int cols) {
  long i = (long)blockIdx.x * 256 + threadIdx.x;
  long n = (long)rows * cols;
  if (i < n) {
    int r = (int)(i / cols), c = (int)(i % cols);
    d[(long)c * rows + r] = (bf16)s[i];
  }
}

// Fused triple-LayerNorm: n1,n2,n3 (bf16) from the same h row (shared mu/var).
__global__ __launch_bounds__(256) void ln3_kernel(
    const float* __restrict__ h,
    const float* __restrict__ g1, const float* __restrict__ b1,
    const float* __restrict__ g2, const float* __restrict__ b2,
    const float* __restrict__ g3, const float* __restrict__ b3,
    bf16* __restrict__ n1, bf16* __restrict__ n2, bf16* __restrict__ n3) {
  __shared__ float tmp[8];
  __shared__ float smu, srs;
  long row = blockIdx.x;
  const float* x = h + row * (long)HDIM;
  float s = 0.f, ss = 0.f;
  for (int i = threadIdx.x; i < HDIM; i += 256) { float v = x[i]; s += v; ss += v * v; }
  int lane = threadIdx.x & 31, wv = threadIdx.x >> 5;
  for (int o = 16; o > 0; o >>= 1) { s += __shfl_down(s, o); ss += __shfl_down(ss, o); }
  if (!lane) tmp[wv] = s;
  __syncthreads();
  if (threadIdx.x == 0) { float a = 0.f; for (int k = 0; k < 8; ++k) a += tmp[k]; smu = a / HDIM; }
  __syncthreads();
  if (!lane) tmp[wv] = ss;
  __syncthreads();
  if (threadIdx.x == 0) {
    float a = 0.f; for (int k = 0; k < 8; ++k) a += tmp[k];
    float var = a / HDIM - smu * smu;
    srs = rsqrtf(var + LNEPS);
  }
  __syncthreads();
  float mu = smu, rstd = srs;
  for (int i = threadIdx.x; i < HDIM; i += 256) {
    float v = (x[i] - mu) * rstd;
    long o = row * (long)HDIM + i;
    n1[o] = (bf16)(v * g1[i] + b1[i]);
    n2[o] = (bf16)(v * g2[i] + b2[i]);
    n3[o] = (bf16)(v * g3[i] + b3[i]);
  }
}

// Gather kv = n1[:, -WKV:, :] into contiguous (BATCH*WKV, HDIM) bf16.
__global__ __launch_bounds__(256) void gather_kv(const bf16* __restrict__ n1,
                                                 bf16* __restrict__ kv) {
  int r = blockIdx.x;                 // 0..BATCH*WKV-1
  int b = r >> 9, sidx = (SEQ - WKV) + (r & 511);
  const bf16* src = n1 + ((long)b * SEQ + sidx) * HDIM;
  bf16* dst = kv + (long)r * HDIM;
  int t = threadIdx.x;                // 256 threads x 4 bf16 (8B)
  *(uint2*)&dst[t * 4] = *(const uint2*)&src[t * 4];
}

// Row softmax f32 -> bf16.
__global__ __launch_bounds__(256) void softmax_rows(const float* __restrict__ S,
                                                    bf16* __restrict__ P, int ncols) {
  __shared__ float tmp[8];
  __shared__ float smx, ssum;
  long row = blockIdx.x;
  const float* x = S + row * (long)ncols;
  int lane = threadIdx.x & 31, wv = threadIdx.x >> 5;
  float mx = -3.0e38f;
  for (int i = threadIdx.x; i < ncols; i += 256) mx = fmaxf(mx, x[i]);
  for (int o = 16; o > 0; o >>= 1) mx = fmaxf(mx, __shfl_down(mx, o));
  if (!lane) tmp[wv] = mx;
  __syncthreads();
  if (threadIdx.x == 0) { float a = -3.0e38f; for (int k = 0; k < 8; ++k) a = fmaxf(a, tmp[k]); smx = a; }
  __syncthreads();
  mx = smx;
  float s = 0.f;
  for (int i = threadIdx.x; i < ncols; i += 256) s += __expf(x[i] - mx);
  for (int o = 16; o > 0; o >>= 1) s += __shfl_down(s, o);
  if (!lane) tmp[wv] = s;
  __syncthreads();
  if (threadIdx.x == 0) { float a = 0.f; for (int k = 0; k < 8; ++k) a += tmp[k]; ssum = a; }
  __syncthreads();
  float inv = 1.0f / ssum;
  bf16* y = P + row * (long)ncols;
  for (int i = threadIdx.x; i < ncols; i += 256) y[i] = (bf16)(__expf(x[i] - mx) * inv);
}

// ------------------------- WMMA tiled GEMM (bf16) --------------------------
// C[Nrows,Ncols] = A[Nrows,K] @ Wt[Ncols,K]^T (+bias). 128x128x64 block tile,
// 8 waves; each wave owns a 64x32 tile = 4x2 wmma fragments, 16 WMMAs/k-tile.
// Dynamic LDS layout: [A0 | B0 | A1 | B1], each 128x64 bf16 (16 KB) = 64 KB.
// Buffer base pointers are computed arithmetically (no pointer arrays over the
// LDS symbol -- those become addrspacecast static initializers ld.lld rejects).
template <int OUT_BF16>
__global__ __launch_bounds__(256) void gemm_bf16_wmma(
    const bf16* __restrict__ A, const bf16* __restrict__ Wt,
    const float* __restrict__ bias, float* __restrict__ Cf, bf16* __restrict__ Cb,
    int Nrows, int Ncols, int K) {
  extern __shared__ bf16 smem[];
  const int TILE = 128 * 64;
  int tid = threadIdx.x, lane = tid & 31, wid = tid >> 5;
  int wm = (wid >> 2) * 64, wn = (wid & 3) * 32;
  long blockM = (long)blockIdx.y * 128;
  long blockN = (long)blockIdx.x * 128;
  const int T = K >> 6;    // number of 64-wide k tiles (all K here are %64==0)

  f32x8 acc[4][2];
#pragma unroll
  for (int mi = 0; mi < 4; ++mi)
#pragma unroll
    for (int ni = 0; ni < 2; ++ni)
      acc[mi][ni] = (f32x8){0.f,0.f,0.f,0.f,0.f,0.f,0.f,0.f};

  auto compute = [&](const bf16* cA, const bf16* cB) {
#pragma unroll
    for (int ks = 0; ks < 64; ks += 32) {
      bf16x16 bfrag[2];
#pragma unroll
      for (int ni = 0; ni < 2; ++ni)
#pragma unroll
        for (int i = 0; i < 16; ++i)
          bfrag[ni][i] = cB[(wn + ni * 16 + (lane & 15)) * 64 + ks + frag_k(i, lane)];
#pragma unroll
      for (int mi = 0; mi < 4; ++mi) {
        bf16x16 afrag;
#pragma unroll
        for (int i = 0; i < 16; ++i)
          afrag[i] = cA[(wm + mi * 16 + (lane & 15)) * 64 + ks + frag_k(i, lane)];
#pragma unroll
        for (int ni = 0; ni < 2; ++ni)
          acc[mi][ni] = wmma_bf16(afrag, bfrag[ni], acc[mi][ni]);
      }
    }
  };

#if HAVE_ASYNC
  // ---- async double-buffered pipeline: copy(t+1) overlaps compute(t) ----
  const bool full = (blockM + 128 <= Nrows) && (blockN + 128 <= Ncols);
  auto issue_tile = [&](int t, int buf) {
    int kk = t << 6;
    bf16* dA = smem + buf * 2 * TILE;        // A_buf
    bf16* dB = dA + TILE;                    // B_buf
#pragma unroll
    for (int j = 0; j < 4; ++j) {
      int c = tid + j * 256, r = c >> 3, c8 = (c & 7) * 8;
      if (blockM + r < Nrows)
        async_cp16(&A[(blockM + r) * (long)K + kk + c8], &dA[r * 64 + c8]);
      if (blockN + r < Ncols)
        async_cp16(&Wt[(blockN + r) * (long)K + kk + c8], &dB[r * 64 + c8]);
    }
  };
  issue_tile(0, 0);
  for (int t = 0; t < T; ++t) {
    int cur = t & 1;
    if (t + 1 < T) {
      issue_tile(t + 1, cur ^ 1);
      if (full) __builtin_amdgcn_s_wait_asynccnt(8);  // tile t done, t+1 in flight
      else      __builtin_amdgcn_s_wait_asynccnt(0);  // partial tile: conservative
    } else {
      __builtin_amdgcn_s_wait_asynccnt(0);
    }
    __syncthreads();
    const bf16* cA = smem + cur * 2 * TILE;
    compute(cA, cA + TILE);
    __syncthreads();  // everyone done reading before tile t+2 overwrites
  }
#else
  // ---- fallback: register-staged pipeline (loads of t+1 overlap compute t) --
  uint4 ra[4], rb[4];
  auto load_regs = [&](int kk) {
#pragma unroll
    for (int j = 0; j < 4; ++j) {
      int c = tid + j * 256, r = c >> 3, c8 = (c & 7) * 8;
      ra[j] = (blockM + r < Nrows) ? *(const uint4*)&A[(blockM + r) * (long)K + kk + c8]
                                   : make_uint4(0u, 0u, 0u, 0u);
      rb[j] = (blockN + r < Ncols) ? *(const uint4*)&Wt[(blockN + r) * (long)K + kk + c8]
                                   : make_uint4(0u, 0u, 0u, 0u);
    }
  };
  load_regs(0);
  for (int t = 0; t < T; ++t) {
#pragma unroll
    for (int j = 0; j < 4; ++j) {
      int c = tid + j * 256, r = c >> 3, c8 = (c & 7) * 8;
      *(uint4*)&smem[r * 64 + c8] = ra[j];
      *(uint4*)&smem[TILE + r * 64 + c8] = rb[j];
    }
    __syncthreads();
    if (t + 1 < T) load_regs((t + 1) << 6);
    compute(smem, smem + TILE);
    __syncthreads();
  }
#endif

  // C 16x16 f32 layout: lane 0-15 -> col=lane, rows r..; lanes 16-31 -> rows +8.
  int rb2 = (lane >> 4) << 3, cn = lane & 15;
#pragma unroll
  for (int mi = 0; mi < 4; ++mi)
#pragma unroll
    for (int ni = 0; ni < 2; ++ni)
#pragma unroll
      for (int r = 0; r < 8; ++r) {
        long row = blockM + wm + mi * 16 + rb2 + r;
        long col = blockN + wn + ni * 16 + cn;
        if (row < Nrows && col < Ncols) {
          float v = acc[mi][ni][r] + (bias ? bias[col] : 0.f);
          if (OUT_BF16) Cb[row * Ncols + col] = (bf16)v;
          else          Cf[row * Ncols + col] = v;
        }
      }
}

// --------------------- fused cross-attention (WMMA) ------------------------
// One wave per (16-query tile, head, batch). Q:16x64, K/V: 512x64 (bf16).
// scores 16x512 f32 live in LDS; K and V tiles share one 16KB LDS buffer.
__global__ __launch_bounds__(32) void attn_fused(const bf16* __restrict__ qb,
                                                 const bf16* __restrict__ kb,
                                                 const bf16* __restrict__ vb,
                                                 bf16* __restrict__ ctx) {
  __shared__ float scr[16 * 512];      // 32 KB
  __shared__ bf16  kvt[128 * DHEAD];   // 16 KB, reused for K then V tiles
  int lane = threadIdx.x;
  int qt = blockIdx.x, head = blockIdx.y, b = blockIdx.z;
  int m = lane & 15;

  // Q fragments (held in registers for the whole kernel)
  bf16x16 qf[2];
  {
    long qrow = (long)b * SEQ + qt * 16 + m;
    const bf16* qp = qb + qrow * HDIM + head * DHEAD;
#pragma unroll
    for (int kf = 0; kf < 2; ++kf)
#pragma unroll
      for (int i = 0; i < 16; ++i) qf[kf][i] = qp[kf * 32 + frag_k(i, lane)];
  }

  // ---- scores = (Q @ K^T) / 8 ----
  for (int kb0 = 0; kb0 < WKV; kb0 += 128) {
    const bf16* ks = kb + ((long)b * WKV + kb0) * HDIM + head * DHEAD;
    for (int c = lane; c < 128 * DHEAD / 8; c += 32) {
      int kr = c >> 3, c8 = (c & 7) * 8;
      *(uint4*)&kvt[kr * DHEAD + c8] = *(const uint4*)&ks[(long)kr * HDIM + c8];
    }
    __syncthreads();
#pragma unroll
    for (int kt = 0; kt < 128; kt += 16) {
      f32x8 acc = (f32x8){0.f,0.f,0.f,0.f,0.f,0.f,0.f,0.f};
#pragma unroll
      for (int kf = 0; kf < 2; ++kf) {
        bf16x16 bfb;
#pragma unroll
        for (int i = 0; i < 16; ++i)
          bfb[i] = kvt[(kt + m) * DHEAD + kf * 32 + frag_k(i, lane)];
        acc = wmma_bf16(qf[kf], bfb, acc);
      }
      int rb = (lane >> 4) << 3;
#pragma unroll
      for (int r = 0; r < 8; ++r)
        scr[(rb + r) * 512 + kb0 + kt + m] = acc[r] * 0.125f;  // 1/sqrt(64)
    }
    __syncthreads();
  }

  // ---- row softmax stats (row = lane&15; halves combined via shfl_xor 16) --
  int half = lane >> 4;
  float mx = -3.0e38f;
  for (int c = half * 256; c < half * 256 + 256; ++c)
    mx = fmaxf(mx, scr[m * 512 + c]);
  mx = fmaxf(mx, __shfl_xor(mx, 16));
  float sum = 0.f;
  for (int c = half * 256; c < half * 256 + 256; ++c)
    sum += __expf(scr[m * 512 + c] - mx);
  sum += __shfl_xor(sum, 16);
  float inv = 1.0f / sum;

  // ---- out = softmax(scores) @ V ----
  f32x8 oacc[4];
#pragma unroll
  for (int nt = 0; nt < 4; ++nt) oacc[nt] = (f32x8){0.f,0.f,0.f,0.f,0.f,0.f,0.f,0.f};

  for (int kb0 = 0; kb0 < WKV; kb0 += 128) {
    const bf16* vs = vb + ((long)b * WKV + kb0) * HDIM + head * DHEAD;
    __syncthreads();
    for (int c = lane; c < 128 * DHEAD / 8; c += 32) {
      int kr = c >> 3, c8 = (c & 7) * 8;
      *(uint4*)&kvt[kr * DHEAD + c8] = *(const uint4*)&vs[(long)kr * HDIM + c8];
    }
    __syncthreads();
#pragma unroll
    for (int ksx = 0; ksx < 128; ksx += 32) {
      bf16x16 af;   // probs fragment: exp folded into the gather
#pragma unroll
      for (int i = 0; i < 16; ++i) {
        int k = kb0 + ksx + frag_k(i, lane);
        af[i] = (bf16)(__expf(scr[m * 512 + k] - mx) * inv);
      }
#pragma unroll
      for (int nt = 0; nt < 4; ++nt) {
        bf16x16 bfb;
#pragma unroll
        for (int i = 0; i < 16; ++i)
          bfb[i] = kvt[(ksx + frag_k(i, lane)) * DHEAD + nt * 16 + m];
        oacc[nt] = wmma_bf16(af, bfb, oacc[nt]);
      }
    }
  }

  long orow = (long)b * SEQ + qt * 16;
  int rb = (lane >> 4) << 3;
#pragma unroll
  for (int nt = 0; nt < 4; ++nt)
#pragma unroll
    for (int r = 0; r < 8; ++r)
      ctx[(orow + rb + r) * HDIM + head * DHEAD + nt * 16 + m] = (bf16)oacc[nt][r];
}

// -------------------- final combine + LayerNorm ----------------------------
__global__ __launch_bounds__(256) void final_combine(
    const float* __restrict__ h, const float* __restrict__ sd,
    const float* __restrict__ ld, const float* __restrict__ pd,
    const float* __restrict__ iw, const float* __restrict__ g,
    const float* __restrict__ bb, float* __restrict__ out) {
  __shared__ float tmp[8];
  __shared__ float smu, srs;
  long row = blockIdx.x;
  float i0 = iw[0], i1 = iw[1], i2 = iw[2];
  float mw = fmaxf(i0, fmaxf(i1, i2));
  float e0 = __expf(i0 - mw), e1 = __expf(i1 - mw), e2 = __expf(i2 - mw);
  float wsum = e0 + e1 + e2;
  float w0 = e0 / wsum, w1 = e1 / wsum, w2 = e2 / wsum;

  long base = row * (long)HDIM;
  float t[4];
  float s = 0.f, ss = 0.f;
#pragma unroll
  for (int j = 0; j < 4; ++j) {
    int i = j * 256 + threadIdx.x;
    float v = h[base + i] + w0 * sd[base + i] + w1 * ld[base + i] + w2 * pd[base + i];
    t[j] = v; s += v; ss += v * v;
  }
  int lane = threadIdx.x & 31, wv = threadIdx.x >> 5;
  for (int o = 16; o > 0; o >>= 1) { s += __shfl_down(s, o); ss += __shfl_down(ss, o); }
  if (!lane) tmp[wv] = s;
  __syncthreads();
  if (threadIdx.x == 0) { float a = 0.f; for (int k = 0; k < 8; ++k) a += tmp[k]; smu = a / HDIM; }
  __syncthreads();
  if (!lane) tmp[wv] = ss;
  __syncthreads();
  if (threadIdx.x == 0) {
    float a = 0.f; for (int k = 0; k < 8; ++k) a += tmp[k];
    srs = rsqrtf(a / HDIM - smu * smu + LNEPS);
  }
  __syncthreads();
  float mu = smu, rstd = srs;
#pragma unroll
  for (int j = 0; j < 4; ++j) {
    int i = j * 256 + threadIdx.x;
    out[base + i] = (t[j] - mu) * rstd * g[i] + bb[i];
  }
}

// ------------------------------ launcher -----------------------------------

static inline void launch_gemm(const bf16* A, const bf16* Wt, const float* bias,
                               void* C, bool out_bf16, int Nr, int Nc, int K,
                               hipStream_t s) {
  dim3 grid((Nc + 127) / 128, (Nr + 127) / 128), block(256);
  const size_t shmem = 4 * 128 * 64 * sizeof(bf16);  // 64 KB (A0,B0,A1,B1)
  if (out_bf16)
    gemm_bf16_wmma<1><<<grid, block, shmem, s>>>(A, Wt, bias, nullptr, (bf16*)C, Nr, Nc, K);
  else
    gemm_bf16_wmma<0><<<grid, block, shmem, s>>>(A, Wt, bias, (float*)C, nullptr, Nr, Nc, K);
}

extern "C" void kernel_launch(void* const* d_in, const int* in_sizes, int n_in,
                              void* d_out, int out_size, void* d_ws, size_t ws_size,
                              hipStream_t stream) {
  (void)in_sizes; (void)n_in; (void)out_size; (void)ws_size;

  const float* h        = (const float*)d_in[0];
  const float* wa_g     = (const float*)d_in[1];
  const float* wa_b     = (const float*)d_in[2];
  const float* attn_in_w  = (const float*)d_in[3];
  const float* attn_in_b  = (const float*)d_in[4];
  const float* attn_out_w = (const float*)d_in[5];
  const float* attn_out_b = (const float*)d_in[6];
  const float* sm_g     = (const float*)d_in[7];
  const float* sm_b     = (const float*)d_in[8];
  const float* sm_q_w   = (const float*)d_in[9];
  const float* sm_q_b   = (const float*)d_in[10];
  const float* sm_out_w = (const float*)d_in[11];
  const float* sm_out_b = (const float*)d_in[12];
  const float* memory   = (const float*)d_in[13];
  const float* pm_g     = (const float*)d_in[14];
  const float* pm_b     = (const float*)d_in[15];
  const float* pers     = (const float*)d_in[16];
  const float* pm_out_w = (const float*)d_in[17];
  const float* pm_out_b = (const float*)d_in[18];
  const float* int_w    = (const float*)d_in[19];
  const float* int_g    = (const float*)d_in[20];
  const float* int_b    = (const float*)d_in[21];

  const size_t HH = (size_t)HDIM * HDIM;
  char* wsp = (char*)d_ws;
  size_t off = 0;
  auto take = [&](size_t bytes) -> char* {
    char* p = wsp + off;
    off = (off + bytes + 255) & ~(size_t)255;
    return p;
  };

  bf16* wq3   = (bf16*)take(3 * HH * 2);                 // q,k,v weights (bf16)
  bf16* wo    = (bf16*)take(HH * 2);
  bf16* wsq   = (bf16*)take(HH * 2);
  bf16* wso   = (bf16*)take(HH * 2);
  bf16* wpo   = (bf16*)take(HH * 2);
  bf16* memB  = (bf16*)take((size_t)MMEM * HDIM * 2);
  bf16* memT  = (bf16*)take((size_t)MMEM * HDIM * 2);    // (HDIM, MMEM)
  bf16* persB = (bf16*)take((size_t)PVEC * HDIM * 2);
  bf16* persT = (bf16*)take((size_t)PVEC * HDIM * 2);    // (HDIM, PVEC)
  bf16* n1    = (bf16*)take((size_t)NTOK * HDIM * 2);
  bf16* n2    = (bf16*)take((size_t)NTOK * HDIM * 2);
  bf16* n3    = (bf16*)take((size_t)NTOK * HDIM * 2);
  bf16* kvn1  = (bf16*)take((size_t)BATCH * WKV * HDIM * 2);
  bf16* qbuf  = (bf16*)take((size_t)NTOK * HDIM * 2);
  bf16* kbuf  = (bf16*)take((size_t)BATCH * WKV * HDIM * 2);
  bf16* vbuf  = (bf16*)take((size_t)BATCH * WKV * HDIM * 2);
  bf16* ctx   = (bf16*)take((size_t)NTOK * HDIM * 2);
  bf16* qx    = (bf16*)take((size_t)NTOK * HDIM * 2);
  bf16* qmem  = (bf16*)take((size_t)NTOK * HDIM * 2);
  bf16* pv    = (bf16*)take((size_t)NTOK * HDIM * 2);
  const int CH = 2048;                                   // row chunk for memory attention
  float* mscore = (float*)take((size_t)CH * MMEM * 4);
  bf16*  mprob  = (bf16*)take((size_t)CH * MMEM * 2);
  float* pscore = (float*)take((size_t)NTOK * PVEC * 4);
  bf16*  pprob  = (bf16*)take((size_t)NTOK * PVEC * 2);
  float* sdelta = (float*)take((size_t)NTOK * HDIM * 4);
  float* ldelta = (float*)take((size_t)NTOK * HDIM * 4);
  float* pdelta = (float*)take((size_t)NTOK * HDIM * 4);

  auto cvt = [&](const float* s, bf16* d, long n) {
    cvt_f32_bf16<<<dim3((unsigned)((n + 255) / 256)), dim3(256), 0, stream>>>(s, d, n);
  };

  // 1) pack weights to bf16 (+ transposes for right-multiplies)
  cvt(attn_in_w, wq3, 3 * (long)HH);
  cvt(attn_out_w, wo, (long)HH);
  cvt(sm_q_w, wsq, (long)HH);
  cvt(sm_out_w, wso, (long)HH);
  cvt(pm_out_w, wpo, (long)HH);
  cvt(memory, memB, (long)MMEM * HDIM);
  cvt(pers, persB, (long)PVEC * HDIM);
  {
    long n = (long)MMEM * HDIM;
    transpose_f32_bf16<<<dim3((unsigned)((n + 255) / 256)), dim3(256), 0, stream>>>(memory, memT, MMEM, HDIM);
    long n2e = (long)PVEC * HDIM;
    transpose_f32_bf16<<<dim3((unsigned)((n2e + 255) / 256)), dim3(256), 0, stream>>>(pers, persT, PVEC, HDIM);
  }

  // 2) fused LN -> n1,n2,n3 ; gather kv slice
  ln3_kernel<<<NTOK, 256, 0, stream>>>(h, wa_g, wa_b, sm_g, sm_b, pm_g, pm_b, n1, n2, n3);
  gather_kv<<<BATCH * WKV, 256, 0, stream>>>(n1, kvn1);

  // 3) short-term attention branch
  launch_gemm(n1,   wq3,            attn_in_b,            qbuf, true, NTOK,        HDIM, HDIM, stream);
  launch_gemm(kvn1, wq3 + HH,       attn_in_b + HDIM,     kbuf, true, BATCH * WKV, HDIM, HDIM, stream);
  launch_gemm(kvn1, wq3 + 2 * HH,   attn_in_b + 2 * HDIM, vbuf, true, BATCH * WKV, HDIM, HDIM, stream);
  attn_fused<<<dim3(SEQ / 16, NHEAD, BATCH), 32, 0, stream>>>(qbuf, kbuf, vbuf, ctx);
  launch_gemm(ctx, wo, attn_out_b, sdelta, false, NTOK, HDIM, HDIM, stream);

  // 4) long-term memory branch (chunked over rows to bound scratch)
  launch_gemm(n2, wsq, sm_q_b, qx, true, NTOK, HDIM, HDIM, stream);
  for (int c = 0; c < NTOK / CH; ++c) {
    const bf16* qxc = qx + (size_t)c * CH * HDIM;
    launch_gemm(qxc, memB, nullptr, mscore, false, CH, MMEM, HDIM, stream);
    softmax_rows<<<CH, 256, 0, stream>>>(mscore, mprob, MMEM);
    launch_gemm(mprob, memT, nullptr, qmem + (size_t)c * CH * HDIM, true, CH, HDIM, MMEM, stream);
  }
  launch_gemm(qmem, wso, sm_out_b, ldelta, false, NTOK, HDIM, HDIM, stream);

  // 5) persistent-memory branch
  launch_gemm(n3, persB, nullptr, pscore, false, NTOK, PVEC, HDIM, stream);
  softmax_rows<<<NTOK, 256, 0, stream>>>(pscore, pprob, PVEC);
  launch_gemm(pprob, persT, nullptr, pv, true, NTOK, HDIM, PVEC, stream);
  launch_gemm(pv, wpo, pm_out_b, pdelta, false, NTOK, HDIM, HDIM, stream);

  // 6) integrate + final LN  (surprise term is multiplied by 0.0 -> skipped)
  final_combine<<<NTOK, 256, 0, stream>>>(h, sdelta, ldelta, pdelta,
                                          int_w, int_g, int_b, (float*)d_out);
}
